// DbrxExperts_36971078484324
// MI455X (gfx1250) — matrix operand order repaired and checked by reading, not compiled, not run on hardware
//
#include <hip/hip_runtime.h>
#include <hip/hip_bf16.h>

#define D_MODEL 2048
#define FFN     4096
#define NEXP    8

typedef __attribute__((ext_vector_type(16))) __bf16 v16bf;
typedef __attribute__((ext_vector_type(8)))  float  v8f;
typedef __attribute__((ext_vector_type(4)))  int    v4i;

union Frag {
    uint4 u[2];
    v16bf bf;
};
union Acc {
    v8f   v;
    float f[8];
};

#define AS1 __attribute__((address_space(1)))
#define AS3 __attribute__((address_space(3)))

#if __has_builtin(__builtin_amdgcn_global_load_async_to_lds_b128)
#define HAVE_ASYNC_LDS 1
static __device__ __forceinline__ void async_copy16(const void* g, void* l) {
    __builtin_amdgcn_global_load_async_to_lds_b128((AS1 v4i*)g, (AS3 v4i*)l, 0, 0);
}
static __device__ __forceinline__ void wait_asynccnt_le1() {
#if __has_builtin(__builtin_amdgcn_s_wait_asynccnt)
    __builtin_amdgcn_s_wait_asynccnt(1);
#else
    asm volatile("s_wait_asynccnt 0x1" ::: "memory");
#endif
}
#else
#define HAVE_ASYNC_LDS 0
#endif

static __device__ __forceinline__ unsigned int f2bf_bits(float f) {
    unsigned int u = __builtin_bit_cast(unsigned int, f);
    u += 0x7FFFu + ((u >> 16) & 1u);   // round-to-nearest-even
    return u >> 16;
}
static __device__ __forceinline__ unsigned int pack2(float a, float b) {
    return f2bf_bits(a) | (f2bf_bits(b) << 16);
}
// silu via v_rcp_f32 (avoids IEEE div_scale/div_fmas expansion)
static __device__ __forceinline__ float silu_fast(float g) {
    return g * __builtin_amdgcn_rcpf(1.0f + __expf(-g));
}

// ---------------------------------------------------------------- zero output
__global__ void zero_kernel(float4* __restrict__ out, int n4) {
    int i = blockIdx.x * blockDim.x + threadIdx.x;
    if (i < n4) out[i] = make_float4(0.f, 0.f, 0.f, 0.f);
}

// ---------------------------------------------------------------- cvt fp32->bf16
__global__ void cvt_bf16_kernel(const float* __restrict__ x,
                                unsigned short* __restrict__ xb, int n4) {
    int i = blockIdx.x * blockDim.x + threadIdx.x;
    if (i >= n4) return;
    float4 f = ((const float4*)x)[i];
    uint2 o;
    o.x = pack2(f.x, f.y);
    o.y = pack2(f.z, f.w);
    ((uint2*)xb)[i] = o;
}

// ---------------------------------------------------------------- router
// one wave32 per token: logits -> softmax(f32) -> top2 -> renorm -> combine row
__global__ __launch_bounds__(256)
void router_kernel(const float* __restrict__ x, const float* __restrict__ rw,
                   float* __restrict__ combine, int T) {
    int wave = threadIdx.x >> 5;
    int lane = threadIdx.x & 31;
    int t = blockIdx.x * 8 + wave;
    if (t >= T) return;
    const float* xt = x + (size_t)t * D_MODEL;

    float logits[NEXP];
#pragma unroll
    for (int e = 0; e < NEXP; ++e) {
        const float* w = rw + e * D_MODEL;
        float acc = 0.f;
        for (int d = lane; d < D_MODEL; d += 32)
            acc = __builtin_fmaf(xt[d], w[d], acc);
#pragma unroll
        for (int off = 16; off > 0; off >>= 1)
            acc += __shfl_xor(acc, off, 32);
        logits[e] = acc;
    }
    float m = logits[0];
#pragma unroll
    for (int e = 1; e < NEXP; ++e) m = fmaxf(m, logits[e]);
    float p[NEXP], s = 0.f;
#pragma unroll
    for (int e = 0; e < NEXP; ++e) { p[e] = __expf(logits[e] - m); s += p[e]; }
    float inv = 1.f / s;
#pragma unroll
    for (int e = 0; e < NEXP; ++e) p[e] *= inv;

    int i1 = 0;
#pragma unroll
    for (int e = 1; e < NEXP; ++e) if (p[e] > p[i1]) i1 = e;  // lowest index on tie
    int i2 = (i1 == 0) ? 1 : 0;
#pragma unroll
    for (int e = 0; e < NEXP; ++e) if (e != i1 && p[e] > p[i2]) i2 = e;
    float wsum = p[i1] + p[i2];
    float w1 = p[i1] / wsum, w2 = p[i2] / wsum;
    if (lane < NEXP) {
        float c = (lane == i1) ? w1 : (lane == i2) ? w2 : 0.f;
        combine[(size_t)t * NEXP + lane] = c;
    }
}

// ---------------------------------------------------------------- compaction
// one wave per expert; stable in-order scan -> deterministic token lists
__global__ __launch_bounds__(256)
void compact_kernel(const float* __restrict__ combine, int* __restrict__ list,
                    int* __restrict__ cnts, int T) {
    int e    = threadIdx.x >> 5;
    int lane = threadIdx.x & 31;
    if (e >= NEXP) return;
    int* le = list + (size_t)e * T;
    int running = 0;
    for (int base = 0; base < T; base += 32) {
        int t = base + lane;
        bool flag = combine[(size_t)t * NEXP + e] > 0.0f;
        unsigned int m32 = (unsigned int)__ballot(flag);
        int pos = running + __popc(m32 & ((1u << lane) - 1u));
        if (flag) le[pos] = t;
        running += __popc(m32);
    }
    if (lane == 0) cnts[e] = running;
}

// ---------------------------------------------------------------- GEMM common
#define MT 64
#define NT 64
#define KT 32
#define LDST 40   // padded LDS row stride (bf16 elems) -> conflict-free b128 reads

// h[pos,:] = silu(x[tok] @ W1^T) * (x[tok] @ V1^T) for gathered tokens
__global__ __launch_bounds__(256)
void gemm1_kernel(const unsigned short* __restrict__ xb,   // [T, D] bf16
                  const float* __restrict__ wsE,           // [2*FFN, D] fp32
                  unsigned short* __restrict__ hb,         // [cnt(pad), FFN] bf16
                  const int* __restrict__ le,              // token list for expert
                  const int* __restrict__ cntp) {
    const int cnt = *cntp;
    const int gm0 = blockIdx.y * MT;
    if (gm0 >= cnt) return;                      // uniform per block
    const int gn0 = blockIdx.x * NT;

#if HAVE_ASYNC_LDS
    __shared__ unsigned short As[2][MT * LDST];  // double-buffered, DMA-fed
#else
    __shared__ unsigned short As[1][MT * LDST];
#endif
    __shared__ unsigned short Bg[NT * LDST];
    __shared__ unsigned short Bu[NT * LDST];

    const int tid   = threadIdx.x;
    const int wave  = tid >> 5, lane = tid & 31;
    const int n_sub = wave & 3, m_pair = wave >> 2;
    const int kh = lane >> 4, hr = lane & 15;

    v8f accg0 = {}, accg1 = {}, accu0 = {}, accu1 = {};

    const int r  = tid >> 2;        // staging row 0..63
    const int cb = (tid & 3) * 8;   // staging col 0,8,16,24

    const int row = gm0 + r;
    const int tok = le[row < cnt ? row : cnt - 1];   // clamp padding rows
    const unsigned short* aSrc = xb  + (size_t)tok * D_MODEL + cb;
    const float*          gSrc = wsE + (size_t)(gn0 + r) * D_MODEL + cb;
    const float*          uSrc = wsE + (size_t)(FFN + gn0 + r) * D_MODEL + cb;

#if HAVE_ASYNC_LDS
    unsigned short* aDst0 = &As[0][r * LDST + cb];
    unsigned short* aDst1 = &As[1][r * LDST + cb];
    async_copy16(aSrc, aDst0);                 // prologue: tile k0=0
#endif

    int buf = 0;
    for (int k0 = 0; k0 < D_MODEL; k0 += KT) {
#if !HAVE_ASYNC_LDS
        uint4 av = *(const uint4*)(aSrc + k0);
#endif
        float4 g0 = *(const float4*)(gSrc + k0);
        float4 g1 = *(const float4*)(gSrc + k0 + 4);
        float4 u0 = *(const float4*)(uSrc + k0);
        float4 u1 = *(const float4*)(uSrc + k0 + 4);
        __builtin_prefetch(gSrc + k0 + KT, 0, 1);
        __builtin_prefetch(uSrc + k0 + KT, 0, 1);

        __syncthreads();                       // prior LDS reads complete
#if HAVE_ASYNC_LDS
        // prefetch next A tile into the idle buffer (wraps harmlessly at end)
        int knext = (k0 + KT < D_MODEL) ? (k0 + KT) : 0;
        async_copy16(aSrc + knext, buf ? (void*)aDst0 : (void*)aDst1);
#else
        *(uint4*)&As[0][r * LDST + cb] = av;
#endif
        uint4 gv, uv;
        gv.x = pack2(g0.x, g0.y); gv.y = pack2(g0.z, g0.w);
        gv.z = pack2(g1.x, g1.y); gv.w = pack2(g1.z, g1.w);
        uv.x = pack2(u0.x, u0.y); uv.y = pack2(u0.z, u0.w);
        uv.z = pack2(u1.x, u1.y); uv.w = pack2(u1.z, u1.w);
        *(uint4*)&Bg[r * LDST + cb] = gv;
        *(uint4*)&Bu[r * LDST + cb] = uv;
#if HAVE_ASYNC_LDS
        wait_asynccnt_le1();                   // oldest (current tile) landed
#endif
        __syncthreads();                       // all waves' writes visible

        Frag a0, a1, bg, bu;
        const int ac = kh * 8;  // A: interleaved half-lane K packing
        a0.u[0] = *(const uint4*)&As[buf][(m_pair * 32 + hr) * LDST + ac];
        a0.u[1] = *(const uint4*)&As[buf][(m_pair * 32 + hr) * LDST + 16 + ac];
        a1.u[0] = *(const uint4*)&As[buf][(m_pair * 32 + 16 + hr) * LDST + ac];
        a1.u[1] = *(const uint4*)&As[buf][(m_pair * 32 + 16 + hr) * LDST + 16 + ac];
        const int bc = kh * 8;  // B: interleaved half-lane K packing
        bg.u[0] = *(const uint4*)&Bg[(n_sub * 16 + hr) * LDST + bc];
        bg.u[1] = *(const uint4*)&Bg[(n_sub * 16 + hr) * LDST + bc + 16];
        bu.u[0] = *(const uint4*)&Bu[(n_sub * 16 + hr) * LDST + bc];
        bu.u[1] = *(const uint4*)&Bu[(n_sub * 16 + hr) * LDST + bc + 16];

        accg0 = __builtin_amdgcn_wmma_f32_16x16x32_bf16(false, a0.bf, false, bg.bf, (short)0, accg0, false, false);
        accg1 = __builtin_amdgcn_wmma_f32_16x16x32_bf16(false, a1.bf, false, bg.bf, (short)0, accg1, false, false);
        accu0 = __builtin_amdgcn_wmma_f32_16x16x32_bf16(false, a0.bf, false, bu.bf, (short)0, accu0, false, false);
        accu1 = __builtin_amdgcn_wmma_f32_16x16x32_bf16(false, a1.bf, false, bu.bf, (short)0, accu1, false, false);
#if HAVE_ASYNC_LDS
        buf ^= 1;
#endif
    }

    Acc g0a, g1a, u0a, u1a;
    g0a.v = accg0; g1a.v = accg1; u0a.v = accu0; u1a.v = accu1;
    const int ncol = gn0 + n_sub * 16 + hr;   // C/D layout: n = lane&15
#pragma unroll
    for (int v = 0; v < 8; ++v) {
        int m0 = gm0 + m_pair * 32 + kh * 8 + v;      // m = v + (lane>=16)*8
        if (m0 < cnt) {
            float h0 = silu_fast(g0a.f[v]) * u0a.f[v];
            hb[(size_t)m0 * FFN + ncol] = (unsigned short)f2bf_bits(h0);
        }
        int m1 = m0 + 16;
        if (m1 < cnt) {
            float h1 = silu_fast(g1a.f[v]) * u1a.f[v];
            hb[(size_t)m1 * FFN + ncol] = (unsigned short)f2bf_bits(h1);
        }
    }
}

// out[tok] += combine[tok,e] * (h[pos] @ W2^T), scattered via token list
__global__ __launch_bounds__(256)
void gemm2_kernel(const unsigned short* __restrict__ hb,   // [cnt(pad), FFN] bf16
                  const float* __restrict__ w2E,           // [D_MODEL, FFN] fp32
                  const float* __restrict__ combine,       // [T, NEXP]
                  float* __restrict__ out,                 // [T, D_MODEL]
                  const int* __restrict__ le,
                  const int* __restrict__ cntp,
                  int expert) {
    const int cnt = *cntp;
    const int gm0 = blockIdx.y * MT;
    if (gm0 >= cnt) return;
    const int gn0 = blockIdx.x * NT;

#if HAVE_ASYNC_LDS
    __shared__ unsigned short As[2][MT * LDST];
#else
    __shared__ unsigned short As[1][MT * LDST];
#endif
    __shared__ unsigned short Bs[NT * LDST];

    const int tid   = threadIdx.x;
    const int wave  = tid >> 5, lane = tid & 31;
    const int n_sub = wave & 3, m_pair = wave >> 2;
    const int kh = lane >> 4, hr = lane & 15;

    v8f acc0 = {}, acc1 = {};

    const int r  = tid >> 2;
    const int cb = (tid & 3) * 8;

    const int arow = (gm0 + r) < cnt ? (gm0 + r) : (cnt - 1);
    const unsigned short* aSrc = hb  + (size_t)arow * FFN + cb;
    const float*          bSrc = w2E + (size_t)(gn0 + r) * FFN + cb;

#if HAVE_ASYNC_LDS
    unsigned short* aDst0 = &As[0][r * LDST + cb];
    unsigned short* aDst1 = &As[1][r * LDST + cb];
    async_copy16(aSrc, aDst0);
#endif

    int buf = 0;
    for (int k0 = 0; k0 < FFN; k0 += KT) {
#if !HAVE_ASYNC_LDS
        uint4 av = *(const uint4*)(aSrc + k0);
#endif
        float4 b0 = *(const float4*)(bSrc + k0);
        float4 b1 = *(const float4*)(bSrc + k0 + 4);
        __builtin_prefetch(bSrc + k0 + KT, 0, 1);

        __syncthreads();
#if HAVE_ASYNC_LDS
        int knext = (k0 + KT < FFN) ? (k0 + KT) : 0;
        async_copy16(aSrc + knext, buf ? (void*)aDst0 : (void*)aDst1);
#else
        *(uint4*)&As[0][r * LDST + cb] = av;
#endif
        uint4 bv;
        bv.x = pack2(b0.x, b0.y); bv.y = pack2(b0.z, b0.w);
        bv.z = pack2(b1.x, b1.y); bv.w = pack2(b1.z, b1.w);
        *(uint4*)&Bs[r * LDST + cb] = bv;
#if HAVE_ASYNC_LDS
        wait_asynccnt_le1();
#endif
        __syncthreads();

        Frag a0, a1, b;
        const int ac = kh * 8;
        a0.u[0] = *(const uint4*)&As[buf][(m_pair * 32 + hr) * LDST + ac];
        a0.u[1] = *(const uint4*)&As[buf][(m_pair * 32 + hr) * LDST + 16 + ac];
        a1.u[0] = *(const uint4*)&As[buf][(m_pair * 32 + 16 + hr) * LDST + ac];
        a1.u[1] = *(const uint4*)&As[buf][(m_pair * 32 + 16 + hr) * LDST + 16 + ac];
        const int bc = kh * 8;
        b.u[0] = *(const uint4*)&Bs[(n_sub * 16 + hr) * LDST + bc];
        b.u[1] = *(const uint4*)&Bs[(n_sub * 16 + hr) * LDST + bc + 16];

        acc0 = __builtin_amdgcn_wmma_f32_16x16x32_bf16(false, a0.bf, false, b.bf, (short)0, acc0, false, false);
        acc1 = __builtin_amdgcn_wmma_f32_16x16x32_bf16(false, a1.bf, false, b.bf, (short)0, acc1, false, false);
#if HAVE_ASYNC_LDS
        buf ^= 1;
#endif
    }

    Acc c0, c1;
    c0.v = acc0; c1.v = acc1;
    const int ncol = gn0 + n_sub * 16 + hr;
#pragma unroll
    for (int v = 0; v < 8; ++v) {
        int m0 = gm0 + m_pair * 32 + kh * 8 + v;
        if (m0 < cnt) {
            int tok = le[m0];
            float cw = combine[(size_t)tok * NEXP + expert];
            size_t i0 = (size_t)tok * D_MODEL + ncol;
            out[i0] += cw * c0.f[v];
        }
        int m1 = m0 + 16;
        if (m1 < cnt) {
            int tok = le[m1];
            float cw = combine[(size_t)tok * NEXP + expert];
            size_t i1 = (size_t)tok * D_MODEL + ncol;
            out[i1] += cw * c1.f[v];
        }
    }
}

// ---------------------------------------------------------------- launcher
extern "C" void kernel_launch(void* const* d_in, const int* in_sizes, int n_in,
                              void* d_out, int out_size, void* d_ws, size_t ws_size,
                              hipStream_t stream) {
    const float* x  = (const float*)d_in[0];   // [B,S,D] fp32
    const float* rw = (const float*)d_in[1];   // [E,D]
    const float* ws = (const float*)d_in[2];   // [E, 2*FFN, D]
    const float* w2 = (const float*)d_in[3];   // [E, D, FFN]
    float* out = (float*)d_out;

    const int T = in_sizes[0] / D_MODEL;       // 4096

    // scratch layout: combine | cnts | list | xb(bf16) | hb(bf16)
    char* p = (char*)d_ws;
    float* combine = (float*)p;              p += (size_t)T * NEXP * sizeof(float);
    int*   cnts    = (int*)p;                p += 64 * sizeof(int);        // padded
    int*   list    = (int*)p;                p += (size_t)NEXP * T * sizeof(int);
    unsigned short* xb = (unsigned short*)p; p += (size_t)T * D_MODEL * sizeof(unsigned short);
    unsigned short* hb = (unsigned short*)p;

    {
        int n4 = T * D_MODEL / 4;  // out is T*D floats
        zero_kernel<<<(n4 + 255) / 256, 256, 0, stream>>>((float4*)out, n4);
        cvt_bf16_kernel<<<(n4 + 255) / 256, 256, 0, stream>>>(x, xb, n4);
    }
    router_kernel<<<(T + 7) / 8, 256, 0, stream>>>(x, rw, combine, T);
    compact_kernel<<<1, 256, 0, stream>>>(combine, list, cnts, T);

    dim3 g1(FFN / NT, T / MT);      // worst-case grid; blocks past cnt exit early
    dim3 g2(D_MODEL / NT, T / MT);
    for (int e = 0; e < NEXP; ++e) {
        const float* wsE = ws + (size_t)e * 2 * FFN * D_MODEL;
        const float* w2E = w2 + (size_t)e * D_MODEL * FFN;
        gemm1_kernel<<<g1, 256, 0, stream>>>(xb, wsE, hb, list + (size_t)e * T, cnts + e);
        gemm2_kernel<<<g2, 256, 0, stream>>>(hb, w2E, combine, out, list + (size_t)e * T, cnts + e, e);
    }
}
